// ResNet_82841329205738
// MI455X (gfx1250) — matrix operand order repaired and checked by reading, not compiled, not run on hardware
//
#include <hip/hip_runtime.h>
#include <hip/hip_bf16.h>

#define B_SAMP      128
#define NUM_CLASSES 12
#define DFEAT       524288ULL   /* 512*32*32 */
#define KCHUNK      4096
#define NKCHUNK     128         /* DFEAT / KCHUNK */

typedef __attribute__((ext_vector_type(16))) __bf16 v16bf;
typedef __attribute__((ext_vector_type(8)))  __bf16 v8bf;
typedef __attribute__((ext_vector_type(8)))  float  v8f;

union Frag16 { v16bf v; v8bf h[2]; };

// ---------------- zero the Gram accumulator ----------------
__global__ void zero_f32_kernel(float* __restrict__ p, int n) {
    int i = blockIdx.x * blockDim.x + threadIdx.x;
    if (i < n) p[i] = 0.0f;
}

// ---------------- fp32 -> bf16 staging (vectorized) ----------------
__global__ void __launch_bounds__(256)
cvt_f32_bf16_kernel(const float* __restrict__ in, __bf16* __restrict__ out) {
    size_t idx = ((size_t)blockIdx.x * 256 + threadIdx.x) * 8;
    const float4 a = *(const float4*)(in + idx);
    const float4 b = *(const float4*)(in + idx + 4);
    v8bf o;
    o[0] = (__bf16)a.x; o[1] = (__bf16)a.y; o[2] = (__bf16)a.z; o[3] = (__bf16)a.w;
    o[4] = (__bf16)b.x; o[5] = (__bf16)b.y; o[6] = (__bf16)b.z; o[7] = (__bf16)b.w;
    *(v8bf*)(out + idx) = o;
}

// ---------------- Gram matrix G = X * X^T via bf16 WMMA ----------------
// One wave per (32x32 tile, 4096-K chunk): 2x2 register blocking -> 2 A-frags
// + 2 B-frags feed 4 WMMAs per k-step (32 B loaded per lane per WMMA instead
// of 64 B), halving L2 fragment traffic. blockIdx.y = K-chunk (slow varying)
// so all 16 tiles of one chunk (1 MB working set) reuse rows out of L2.
// fp32 atomics merge the K-split partials.
__global__ void __launch_bounds__(32)
gram_bf16_wmma_kernel(const __bf16* __restrict__ xb, float* __restrict__ G) {
    const int lane  = threadIdx.x;     // 0..31
    const int hi    = lane >> 4;       // half-wave selector
    const int lo    = lane & 15;
    const int tile  = blockIdx.x;      // 0..15
    const int rbase = (tile >> 2) * 32;
    const int cbase = (tile & 3) * 32;
    const size_t k0 = (size_t)blockIdx.y * KCHUNK;

    // A fragment (16x32 bf16): lane half selects K sub-range {0..7,16..23} vs
    // {8..15,24..31}; two contiguous 16-byte segments per lane.
    const __bf16* pa0 = xb + (size_t)(rbase + lo) * DFEAT + k0 + (size_t)hi * 8;
    const __bf16* pa1 = pa0 + (size_t)16 * DFEAT;
    // B fragment (32x16 bf16): lane lo = column, halves hold K {0..7,16..23} vs
    // {8..15,24..31}; two 16-byte segments per lane (B[k][n] = X[n][k] for the Gram).
    const __bf16* pb0 = xb + (size_t)(cbase + lo) * DFEAT + k0 + (size_t)hi * 8;
    const __bf16* pb1 = pb0 + (size_t)16 * DFEAT;

    v8f acc00 = {}, acc01 = {}, acc10 = {}, acc11 = {};
    #pragma unroll 2
    for (int k = 0; k < KCHUNK; k += 32) {
        Frag16 A0, A1, B0, B1;
        A0.h[0] = *(const v8bf*)(pa0 + k);  A0.h[1] = *(const v8bf*)(pa0 + k + 16);
        A1.h[0] = *(const v8bf*)(pa1 + k);  A1.h[1] = *(const v8bf*)(pa1 + k + 16);
        B0.h[0] = *(const v8bf*)(pb0 + k);  B0.h[1] = *(const v8bf*)(pb0 + k + 16);
        B1.h[0] = *(const v8bf*)(pb1 + k);  B1.h[1] = *(const v8bf*)(pb1 + k + 16);
        __builtin_prefetch(pa0 + k + 1024, 0, 1);   // global_prefetch_b8, ~32 iters ahead
        __builtin_prefetch(pa1 + k + 1024, 0, 1);
        __builtin_prefetch(pb0 + k + 1024, 0, 1);
        __builtin_prefetch(pb1 + k + 1024, 0, 1);
        // (neg_a, A, neg_b, B, c_mod, C, reuse_a, reuse_b)
        acc00 = __builtin_amdgcn_wmma_f32_16x16x32_bf16(
                    false, A0.v, false, B0.v, (short)0, acc00, false, false);
        acc01 = __builtin_amdgcn_wmma_f32_16x16x32_bf16(
                    false, A0.v, false, B1.v, (short)0, acc01, false, false);
        acc10 = __builtin_amdgcn_wmma_f32_16x16x32_bf16(
                    false, A1.v, false, B0.v, (short)0, acc10, false, false);
        acc11 = __builtin_amdgcn_wmma_f32_16x16x32_bf16(
                    false, A1.v, false, B1.v, (short)0, acc11, false, false);
    }

    // C/D layout: VGPR r -> M = base + r + hi*8, N = cbase + lo
    const int gn0 = cbase + lo;
    const int gn1 = gn0 + 16;
    #pragma unroll
    for (int r = 0; r < 8; ++r) {
        const int gm0 = rbase + r + hi * 8;
        const int gm1 = gm0 + 16;
        atomicAdd(&G[gm0 * B_SAMP + gn0], acc00[r]);
        atomicAdd(&G[gm0 * B_SAMP + gn1], acc01[r]);
        atomicAdd(&G[gm1 * B_SAMP + gn0], acc10[r]);
        atomicAdd(&G[gm1 * B_SAMP + gn1], acc11[r]);
    }
}

// ---------------- per-sample weights from Gram matrix ----------------
__global__ void __launch_bounds__(128)
weights_kernel(const float* __restrict__ G, const int* __restrict__ labels,
               float* __restrict__ w) {
    __shared__ float sq[B_SAMP];
    __shared__ int   lab[B_SAMP];
    __shared__ float red[B_SAMP];
    const int i = threadIdx.x;
    sq[i]  = G[i * B_SAMP + i];   // ||x_i||^2 from diag -> exact 0 on diagonal
    lab[i] = labels[i];
    __syncthreads();

    const int li = lab[i];
    float dsum = 0.0f;
    for (int j = 0; j < B_SAMP; ++j) {
        if (lab[j] == li) {
            float d2 = sq[i] + sq[j] - 2.0f * G[i * B_SAMP + j];
            d2 = fmaxf(d2, 0.0f);
            dsum += (d2 > 0.0f) ? sqrtf(d2) : 0.0f;
        }
    }
    const float wi = 1.0f / (sqrtf(fmaf(dsum, dsum, 25.0f)) + 1e-12f);
    red[i] = wi;
    __syncthreads();
    for (int s = 64; s > 0; s >>= 1) {
        if (i < s) red[i] += red[i + s];
        __syncthreads();
    }
    w[i] = wi / (red[0] + 1e-12f);
}

// ---------------- weighted label-grouped scatter-sum ----------------
// grid = (DFEAT/1024, NUM_CLASSES); label test is wave-uniform (scalar branch),
// so each feature element is loaded exactly once across the whole grid.
__global__ void __launch_bounds__(256)
scatter_kernel(const float* __restrict__ feat, const int* __restrict__ labels,
               const float* __restrict__ w, float* __restrict__ out) {
    const int    c = blockIdx.y;
    const size_t p = ((size_t)blockIdx.x * 256 + threadIdx.x) * 4;
    float4 acc = make_float4(0.0f, 0.0f, 0.0f, 0.0f);
    for (int i = 0; i < B_SAMP; ++i) {
        if (labels[i] == c) {                    // scalar (uniform) condition
            const float  wi = w[i];
            const float4 f  = *(const float4*)(feat + (size_t)i * DFEAT + p);
            acc.x = fmaf(wi, f.x, acc.x);
            acc.y = fmaf(wi, f.y, acc.y);
            acc.z = fmaf(wi, f.z, acc.z);
            acc.w = fmaf(wi, f.w, acc.w);
        }
    }
    *(float4*)(out + (size_t)c * DFEAT + p) = acc;
}

extern "C" void kernel_launch(void* const* d_in, const int* in_sizes, int n_in,
                              void* d_out, int out_size, void* d_ws, size_t ws_size,
                              hipStream_t stream) {
    (void)in_sizes; (void)n_in; (void)out_size; (void)ws_size;
    const float* feat   = (const float*)d_in[0];
    const int*   labels = (const int*)d_in[1];
    float*       out    = (float*)d_out;

    // Workspace layout: [bf16 copy of X : 128 MB][G : 64 KB][w : 512 B]
    char*   ws      = (char*)d_ws;
    __bf16* xb      = (__bf16*)ws;
    const size_t bfBytes = (size_t)B_SAMP * DFEAT * sizeof(__bf16);
    float*  G = (float*)(ws + bfBytes);
    float*  w = (float*)(ws + bfBytes + (size_t)B_SAMP * B_SAMP * sizeof(float));

    // 1) zero Gram accumulator (K-split uses atomics)
    zero_f32_kernel<<<(B_SAMP * B_SAMP + 255) / 256, 256, 0, stream>>>(G, B_SAMP * B_SAMP);

    // 2) stage X in bf16 (8 elems/thread)
    const size_t nElem = (size_t)B_SAMP * DFEAT;          // 67,108,864
    cvt_f32_bf16_kernel<<<(unsigned)(nElem / (256 * 8)), 256, 0, stream>>>(feat, xb);

    // 3) Gram GEMM: 16 (32x32) tiles x 128 K-chunks, one wave each
    dim3 gg(16, NKCHUNK);
    gram_bf16_wmma_kernel<<<gg, 32, 0, stream>>>(xb, G);

    // 4) weights (single workgroup)
    weights_kernel<<<1, 128, 0, stream>>>(G, labels, w);

    // 5) weighted scatter into [12, 512, 32, 32]
    dim3 gs((unsigned)(DFEAT / (256 * 4)), NUM_CLASSES);
    scatter_kernel<<<gs, 256, 0, stream>>>(feat, labels, w, out);
}
